// SA_ConvLSTMCell_75823352644194
// MI455X (gfx1250) — compile-verified, matrix operand order checked by reading, not run on hardware
//
#include <hip/hip_runtime.h>

typedef __attribute__((ext_vector_type(16))) __bf16 v16bf;
typedef __attribute__((ext_vector_type(8)))  __bf16 v8bf;
typedef __attribute__((ext_vector_type(8)))  float  v8f;
typedef __attribute__((ext_vector_type(4)))  unsigned int v4u;
typedef __attribute__((ext_vector_type(8)))  int v8i;
typedef __attribute__((ext_vector_type(4)))  int v4i;

#define HWSZ 4096
#define NTOT (4*64*4096)
#define PADW 68                       // 64 + 2-pixel halo each side

#if defined(__has_builtin)
#if __has_builtin(__builtin_amdgcn_tensor_load_to_lds)
#define HAVE_TDM 1
#endif
#endif
#ifndef HAVE_TDM
#define HAVE_TDM 0
#endif

static __device__ __forceinline__ v8f wmma_bf16(v16bf a, v16bf b, v8f c) {
  return __builtin_amdgcn_wmma_f32_16x16x32_bf16(false, a, false, b, (short)0, c,
                                                 false, false);
}
static __device__ __forceinline__ v16bf cat8(v8bf lo, v8bf hi) {
  return __builtin_shufflevector(lo, hi, 0,1,2,3,4,5,6,7,8,9,10,11,12,13,14,15);
}
static __device__ __forceinline__ float sigf(float x) {
  return 1.0f / (1.0f + __expf(-x));
}

// ---------------------------------------------------------------------------
// Weight packers: bf16 in exact WMMA A-fragment order (one 32B read/k-step).
// ---------------------------------------------------------------------------
__global__ __launch_bounds__(256) void k_pack_w(
    const float* __restrict__ W, __bf16* __restrict__ P, int K)
{
  const int idx  = blockIdx.x * 256 + threadIdx.x;   // (M/16)*(K/32)*512
  const int j    = idx & 15;
  const int lane = (idx >> 4) & 31;
  const int blk  = idx >> 9;                         // mb*(K/32)+kb
  const int kb   = blk % (K / 32);
  const int mb   = blk / (K / 32);
  const int m    = mb * 16 + (lane & 15);
  const int kl   = ((lane >> 4) << 3) + (j < 8 ? j : j + 8);
  P[idx] = (__bf16)W[(size_t)m * K + kb * 32 + kl];
}

// Conv weights: K = 3200 tap-major (kb: tap = kb>>2, ci = (kb&3)*32+klocal).
__global__ __launch_bounds__(256) void k_pack_conv(
    const float* __restrict__ W, __bf16* __restrict__ P)
{
  const int idx  = blockIdx.x * 256 + threadIdx.x;   // 16*100*512 = 819200
  const int j    = idx & 15;
  const int lane = (idx >> 4) & 31;
  const int blk  = idx >> 9;
  const int kb   = blk % 100;
  const int mb   = blk / 100;
  const int m    = mb * 16 + (lane & 15);
  const int kl   = ((lane >> 4) << 3) + (j < 8 ? j : j + 8);
  const int ci   = ((kb & 3) << 5) + kl;
  const int tap  = kb >> 2;
  P[idx] = (__bf16)W[((size_t)m * 128 + ci) * 25 + tap];
}

// concat(x,h) -> bf16 with 2-pixel zero halo: [b][68][68][128ci].
__global__ __launch_bounds__(256) void k_xh_pad(
    const float* __restrict__ x, const float* __restrict__ h,
    __bf16* __restrict__ xhp)
{
  const size_t idx = (size_t)blockIdx.x * 256 + threadIdx.x;  // 4*68*68*128
  const int    ci  = idx & 127;
  const size_t t   = idx >> 7;
  const int    col = (int)(t % PADW);
  const size_t t2  = t / PADW;
  const int    row = (int)(t2 % PADW);
  const int    b   = (int)(t2 / PADW);
  const int yy = row - 2, xx = col - 2;
  float v = 0.0f;
  if (yy >= 0 && yy < 64 && xx >= 0 && xx < 64) {
    const float* src = (ci < 64) ? (x + ((size_t)b * 64 + ci) * HWSZ)
                                 : (h + ((size_t)b * 64 + (ci - 64)) * HWSZ);
    v = src[yy * 64 + xx];
  }
  xhp[idx] = (__bf16)v;
}

// [B][64][HW] f32 -> [B][n][64] bf16 (pixel-major)
__global__ __launch_bounds__(256) void k_t64(
    const float* __restrict__ src, __bf16* __restrict__ dst)
{
  const size_t idx = (size_t)blockIdx.x * 256 + threadIdx.x;  // B*HW*64
  const int    ch  = idx & 63;
  const size_t n   = (idx >> 6) & 4095;
  const size_t b   = idx >> 18;
  dst[idx] = (__bf16)src[(b * 64 + ch) * HWSZ + n];
}

// ---------------------------------------------------------------------------
// 5x5 conv as implicit GEMM on the padded bf16 input.  16(M) x 128(N) tile
// per wave; no predication; immediate-offset b128 loads only.
// ---------------------------------------------------------------------------
__global__ __launch_bounds__(32) void k_conv_gates(
    const __bf16* __restrict__ xhp, const __bf16* __restrict__ wpack,
    const float* __restrict__ bias, float* __restrict__ cc)
{
  const int lane = threadIdx.x;
  const int q    = lane & 15;
  const int kb8  = (lane >> 4) << 3;
  const int nb   = blockIdx.x;   // 2 image rows
  const int mb   = blockIdx.y;
  const int b    = blockIdx.z;
  const int y0   = nb * 2;

  const __bf16* xb   = xhp + (size_t)b * PADW * PADW * 128;
  const __bf16* base = xb + ((size_t)(y0 + 2) * PADW + (q + 2)) * 128 + kb8;
  const __bf16* wp   = wpack + (size_t)mb * 100 * 512 + lane * 16;

  v8f acc[8] = {};
  for (int tap = 0; tap < 25; ++tap) {
    const int dy = tap / 5 - 2, dx = tap % 5 - 2;
    const __bf16* bt = base + ((long)dy * PADW + dx) * 128;
#pragma unroll
    for (int cb4 = 0; cb4 < 4; ++cb4) {
      const v16bf a = *(const v16bf*)(wp + (size_t)(tap * 4 + cb4) * 512);
#pragma unroll
      for (int s = 0; s < 8; ++s) {
        const __bf16* p = bt + ((s >> 2) * PADW + (s & 3) * 16) * 128 + cb4 * 32;
        acc[s] = wmma_bf16(a, cat8(*(const v8bf*)p, *(const v8bf*)(p + 16)),
                           acc[s]);
      }
    }
  }

  const int hi8 = kb8;
#pragma unroll
  for (int s = 0; s < 8; ++s) {
    const int yy = y0 + (s >> 2);
    const int xx = (s & 3) * 16 + q;
#pragma unroll
    for (int r = 0; r < 8; ++r) {
      const int mrow = mb * 16 + r + hi8;
      cc[((size_t)b * 256 + mrow) * HWSZ + yy * 64 + xx] = acc[s][r] + bias[mrow];
    }
  }
}

// ---------------------------------------------------------------------------
// GroupNorm (groups == channels) on f32 cc planes.
// ---------------------------------------------------------------------------
__global__ __launch_bounds__(256) void k_groupnorm(
    float* __restrict__ cc, const float* __restrict__ g,
    const float* __restrict__ bta)
{
  __shared__ float ssum[256], ssq[256];
  const int tid = threadIdx.x;
  const int chn = blockIdx.x & 255;
  float* p = cc + (size_t)blockIdx.x * HWSZ;

  float s = 0.f, s2 = 0.f;
  for (int i = tid; i < HWSZ; i += 256) { float v = p[i]; s += v; s2 += v * v; }
  ssum[tid] = s; ssq[tid] = s2;
  __syncthreads();
  for (int off = 128; off > 0; off >>= 1) {
    if (tid < off) { ssum[tid] += ssum[tid + off]; ssq[tid] += ssq[tid + off]; }
    __syncthreads();
  }
  const float mu  = ssum[0] * (1.0f / HWSZ);
  const float var = ssq[0] * (1.0f / HWSZ) - mu * mu;
  const float inv = rsqrtf(var + 1e-5f) * g[chn];
  const float bb  = bta[chn];
  for (int i = tid; i < HWSZ; i += 256) p[i] = (p[i] - mu) * inv + bb;
}

// ---------------------------------------------------------------------------
// LSTM gates.
// ---------------------------------------------------------------------------
__global__ __launch_bounds__(256) void k_gates(
    const float* __restrict__ cc, const float* __restrict__ cprev,
    float* __restrict__ c_out, __bf16* __restrict__ hmid_t,
    __bf16* __restrict__ cat2)
{
  const size_t idx = (size_t)blockIdx.x * 256 + threadIdx.x;  // B*HW*64
  const int    ch  = idx & 63;
  const size_t hw  = (idx >> 6) & 4095;
  const size_t b   = idx >> 18;
  const size_t base = b * 256 * HWSZ;
  const size_t cidx = (b * 64 + ch) * HWSZ + hw;

  const float iv = sigf(cc[base + (size_t)ch * HWSZ + hw]);
  const float fv = sigf(cc[base + (size_t)(64 + ch) * HWSZ + hw]);
  const float ov = sigf(cc[base + (size_t)(128 + ch) * HWSZ + hw]);
  const float gv = tanhf(cc[base + (size_t)(192 + ch) * HWSZ + hw]);
  const float cn = fv * cprev[cidx] + iv * gv;
  const float hm = ov * tanhf(cn);
  c_out[cidx]  = cn;
  hmid_t[idx]  = (__bf16)hm;
  cat2[(b * HWSZ + hw) * 192 + 128 + ch] = (__bf16)hm;
}

// ---------------------------------------------------------------------------
// GEMM, Y pixel-major [b][n][ldy], X [b][n][K] bf16, A pre-packed.
// ---------------------------------------------------------------------------
__global__ __launch_bounds__(32) void k_gemm_nm(
    const __bf16* __restrict__ Apack, const float* __restrict__ bias,
    const __bf16* __restrict__ X, __bf16* __restrict__ Y,
    int K, int ldy, long sXb, long sYb)
{
  const int lane = threadIdx.x;
  const int q    = lane & 15;
  const int kb8  = (lane >> 4) << 3;
  const int nb   = blockIdx.x;
  const int mb   = blockIdx.y;
  const __bf16* Xb = X + (size_t)blockIdx.z * sXb;
  __bf16*       Yb = Y + (size_t)blockIdx.z * sYb;
  const int nkb = K >> 5;
  const __bf16* ap   = Apack + (size_t)mb * nkb * 512 + lane * 16;
  const __bf16* xrow = Xb + (size_t)(nb * 128 + q) * K + kb8;

  v8f acc[8] = {};
  for (int kb = 0; kb < nkb; ++kb) {
    const v16bf a = *(const v16bf*)(ap + (size_t)kb * 512);
#pragma unroll
    for (int s = 0; s < 8; ++s) {
      const __bf16* p = xrow + (size_t)s * 16 * K;
      acc[s] = wmma_bf16(a, cat8(*(const v8bf*)p, *(const v8bf*)(p + 16)),
                         acc[s]);
    }
    xrow += 32;
  }
  const int hi8 = kb8;
#pragma unroll
  for (int s = 0; s < 8; ++s) {
    v8bf o;
#pragma unroll
    for (int r = 0; r < 8; ++r)
      o[r] = (__bf16)(acc[s][r] + bias[mb * 16 + r + hi8]);
    *(v8bf*)(Yb + (size_t)(nb * 128 + s * 16 + q) * ldy + mb * 16 + hi8) = o;
  }
}

// Same GEMM but Y channel-major [b][M][HW] (V layout for flash B-frags).
__global__ __launch_bounds__(32) void k_gemm_cm(
    const __bf16* __restrict__ Apack, const float* __restrict__ bias,
    const __bf16* __restrict__ X, __bf16* __restrict__ Y,
    int K, long sXb, long sYb)
{
  const int lane = threadIdx.x;
  const int q    = lane & 15;
  const int kb8  = (lane >> 4) << 3;
  const int nb   = blockIdx.x;
  const int mb   = blockIdx.y;
  const __bf16* Xb = X + (size_t)blockIdx.z * sXb;
  __bf16*       Yb = Y + (size_t)blockIdx.z * sYb;
  const int nkb = K >> 5;
  const __bf16* ap   = Apack + (size_t)mb * nkb * 512 + lane * 16;
  const __bf16* xrow = Xb + (size_t)(nb * 128 + q) * K + kb8;

  v8f acc[8] = {};
  for (int kb = 0; kb < nkb; ++kb) {
    const v16bf a = *(const v16bf*)(ap + (size_t)kb * 512);
#pragma unroll
    for (int s = 0; s < 8; ++s) {
      const __bf16* p = xrow + (size_t)s * 16 * K;
      acc[s] = wmma_bf16(a, cat8(*(const v8bf*)p, *(const v8bf*)(p + 16)),
                         acc[s]);
    }
    xrow += 32;
  }
  const int hi8 = kb8;
#pragma unroll
  for (int s = 0; s < 8; ++s)
#pragma unroll
    for (int r = 0; r < 8; ++r) {
      const int mrow = mb * 16 + r + hi8;
      Yb[(size_t)mrow * HWSZ + nb * 128 + s * 16 + q] =
          (__bf16)(acc[s][r] + bias[mrow]);
    }
}

// ---------------------------------------------------------------------------
// Flash attention, 8 waves per workgroup sharing K/V tiles in LDS.
// Wave 0 stages the tiles with the Tensor Data Mover (2-D descriptor for V,
// 1-D for K) and s_wait_tensorcnt; all waves read fragments via ds_load_b128.
// Falls back to cooperative vector-load staging if the TDM builtin is absent.
// ---------------------------------------------------------------------------
__global__ __launch_bounds__(256) void k_flash(
    const __bf16* __restrict__ Qt, const __bf16* __restrict__ Kht,
    const __bf16* __restrict__ Vhc, const __bf16* __restrict__ Kmt,
    const __bf16* __restrict__ Vmc, __bf16* __restrict__ Zcat)
{
  __shared__ __align__(16) __bf16 sK[32 * 16];   // [key][d]
  __shared__ __align__(16) __bf16 sV[64 * 32];   // [ch][key]

  const int tid  = threadIdx.x;
  const int lane = tid & 31;
  const int wv   = tid >> 5;
  const int q    = lane & 15;
  const int kb8  = (lane >> 4) << 3;
  const int qb   = blockIdx.x * 8 + wv;          // query block of 16
  const int wch  = blockIdx.y;
  const int b    = blockIdx.z;

  const __bf16* Qp = Qt + (size_t)b * HWSZ * 16;
  const __bf16* Kp = (wch ? Kmt : Kht) + (size_t)b * HWSZ * 16;
  const __bf16* Vp = (wch ? Vmc : Vhc) + (size_t)b * 64 * HWSZ;
  __bf16* Zp = Zcat + (size_t)b * HWSZ * 128 + wch * 64;

  const v8bf zv = {};
  const int qcol = qb * 16 + q;
  const v16bf qf = cat8(*(const v8bf*)(Qp + (size_t)qcol * 16 + kb8), zv);

  float rmax = -3.0e38f, rsum = 0.0f;
  v8f acc[4] = {};

  for (int k0 = 0; k0 < HWSZ; k0 += 32) {
    __syncthreads();                 // previous tiles fully consumed
#if HAVE_TDM
    if (wv == 0) {
      const unsigned ldsK = (unsigned)(uintptr_t)&sK[0];
      const unsigned ldsV = (unsigned)(uintptr_t)&sV[0];
      const unsigned long long gaK =
          (unsigned long long)(uintptr_t)(Kp + (size_t)k0 * 16);
      const unsigned long long gaV =
          (unsigned long long)(uintptr_t)(Vp + k0);
      // D# group0: count=1 | lds_addr | global_addr | type=2 ("image")
      const v4u g0K = {1u, ldsK, (unsigned)gaK,
                       (unsigned)(gaK >> 32) | (2u << 30)};
      const v4u g0V = {1u, ldsV, (unsigned)gaV,
                       (unsigned)(gaV >> 32) | (2u << 30)};
      // D# group1: data_size=2B; K: 1-D tile of 512 elems;
      // V: 2-D tile 32(keys) x 64(ch), tensor_dim0_stride = HWSZ.
      const v8i g1K = {0x00010000, (int)(512u << 16), 0, (int)(512u << 16),
                       0, 512, 0, 0};
      const v8i g1V = {0x00010000, (int)((unsigned)HWSZ << 16), 64 << 16,
                       32 << 16, 64, HWSZ, 0, 0};
      const v4i z4 = {};
#if __clang_major__ >= 23
      const v8i z8 = {};
      __builtin_amdgcn_tensor_load_to_lds(g0K, g1K, z4, z4, z8, 0);
      __builtin_amdgcn_tensor_load_to_lds(g0V, g1V, z4, z4, z8, 0);
#else
      __builtin_amdgcn_tensor_load_to_lds(g0K, g1K, z4, z4, 0);
      __builtin_amdgcn_tensor_load_to_lds(g0V, g1V, z4, z4, 0);
#endif
      __builtin_amdgcn_s_wait_tensorcnt(0);
    }
#else
    {
      // cooperative staging: K = 256 dwords, V = 1024 dwords
      unsigned*       lk = (unsigned*)sK;
      const unsigned* gk = (const unsigned*)(Kp + (size_t)k0 * 16);
      lk[tid] = gk[tid];
      unsigned* lv = (unsigned*)sV;
#pragma unroll
      for (int r = 0; r < 4; ++r) {
        const int d  = tid + r * 256;
        const int ch = d >> 4, kp = d & 15;
        lv[d] = *(const unsigned*)(Vp + (size_t)ch * HWSZ + k0 + kp * 2);
      }
    }
#endif
    __syncthreads();                 // tiles ready

    const v16bf ka0 = cat8(*(const v8bf*)(sK + q * 16 + kb8), zv);
    const v16bf ka1 = cat8(*(const v8bf*)(sK + (q + 16) * 16 + kb8), zv);
    v8f s0 = {}, s1 = {};
    s0 = wmma_bf16(ka0, qf, s0);
    s1 = wmma_bf16(ka1, qf, s1);

    float bmax = -3.0e38f;
#pragma unroll
    for (int r = 0; r < 8; ++r) bmax = fmaxf(bmax, fmaxf(s0[r], s1[r]));
    bmax = fmaxf(bmax, __shfl_xor(bmax, 16, 32));

    const float newmax = fmaxf(rmax, bmax);
    const float alpha  = __expf(rmax - newmax);

    v16bf pa; float bsum = 0.0f;
#pragma unroll
    for (int r = 0; r < 8; ++r) {
      const float e0 = __expf(s0[r] - newmax);
      const float e1 = __expf(s1[r] - newmax);
      pa[r]     = (__bf16)e0;
      pa[r + 8] = (__bf16)e1;
      bsum += e0 + e1;
    }
    bsum += __shfl_xor(bsum, 16, 32);
    rsum = rsum * alpha + bsum;
    rmax = newmax;

    float arow[8];
#pragma unroll
    for (int r = 0; r < 8; ++r) arow[r] = __shfl(alpha, r + kb8, 32);

#pragma unroll
    for (int s = 0; s < 4; ++s) {
      const __bf16* vp = sV + (s * 16 + q) * 32 + kb8;
      const v16bf vf = cat8(*(const v8bf*)vp, *(const v8bf*)(vp + 16));
#pragma unroll
      for (int r = 0; r < 8; ++r) acc[s][r] *= arow[r];
      acc[s] = wmma_bf16(pa, vf, acc[s]);
    }
  }

  const float rinv = 1.0f / rsum;
  float lrow[8];
#pragma unroll
  for (int r = 0; r < 8; ++r) lrow[r] = __shfl(rinv, r + kb8, 32);
#pragma unroll
  for (int s = 0; s < 4; ++s)
#pragma unroll
    for (int r = 0; r < 8; ++r)
      Zp[(size_t)(qb * 16 + r + kb8) * 128 + s * 16 + q] =
          (__bf16)(acc[s][r] * lrow[r]);
}

// ---------------------------------------------------------------------------
// Final memory-cell update from comb [b][n][192] bf16.
// ---------------------------------------------------------------------------
__global__ __launch_bounds__(256) void k_mem_out(
    const __bf16* __restrict__ comb, const float* __restrict__ mprev,
    float* __restrict__ out)
{
  const size_t idx = (size_t)blockIdx.x * 256 + threadIdx.x;  // B*HW*64
  const int    ch  = idx & 63;
  const size_t hw  = (idx >> 6) & 4095;
  const size_t b   = idx >> 18;
  const size_t row = (b * HWSZ + hw) * 192;
  const size_t oix = (b * 64 + ch) * HWSZ + hw;

  const float mo = (float)comb[row + ch];
  const float mg = (float)comb[row + 64 + ch];
  const float mi = (float)comb[row + 128 + ch];
  const float si = sigf(mi);
  const float nm = (1.0f - si) * mprev[oix] + si * tanhf(mg);
  out[oix] = sigf(mo) * nm;                 // h_next
  out[2 * (size_t)NTOT + oix] = nm;         // m_next
}

// ---------------------------------------------------------------------------
extern "C" void kernel_launch(void* const* d_in, const int* in_sizes, int n_in,
                              void* d_out, int out_size, void* d_ws, size_t ws_size,
                              hipStream_t stream)
{
  (void)in_sizes; (void)n_in; (void)out_size; (void)ws_size;

  const float* x   = (const float*)d_in[0];
  const float* h   = (const float*)d_in[1];
  const float* c   = (const float*)d_in[2];
  const float* m   = (const float*)d_in[3];
  const float* cw  = (const float*)d_in[4];
  const float* cb  = (const float*)d_in[5];
  const float* gng = (const float*)d_in[6];
  const float* gnb = (const float*)d_in[7];
  const float* wq  = (const float*)d_in[8];
  const float* bq  = (const float*)d_in[9];
  const float* wk  = (const float*)d_in[10];
  const float* bk  = (const float*)d_in[11];
  const float* wv  = (const float*)d_in[12];
  const float* bv  = (const float*)d_in[13];
  const float* wk2 = (const float*)d_in[14];
  const float* bk2 = (const float*)d_in[15];
  const float* wv2 = (const float*)d_in[16];
  const float* bv2 = (const float*)d_in[17];
  const float* wz  = (const float*)d_in[18];
  const float* bz  = (const float*)d_in[19];
  const float* wmw = (const float*)d_in[20];
  const float* bm  = (const float*)d_in[21];
  float* out = (float*)d_out;

  char* wsb = (char*)d_ws;
  auto takeF = [&](size_t nelem) { float*  p = (float*)wsb;  wsb += ((nelem*4 + 255) & ~(size_t)255); return p; };
  auto takeB = [&](size_t nelem) { __bf16* p = (__bf16*)wsb; wsb += ((nelem*2 + 255) & ~(size_t)255); return p; };

  float*  cc     = takeF((size_t)4 * 256 * HWSZ);
  __bf16* xhp    = takeB((size_t)4 * PADW * PADW * 128);
  __bf16* wpc    = takeB((size_t)16 * 100 * 512);
  __bf16* pwq    = takeB((size_t)16 * 64);
  __bf16* pwk    = takeB((size_t)16 * 64);
  __bf16* pwv    = takeB((size_t)64 * 64);
  __bf16* pwk2   = takeB((size_t)16 * 64);
  __bf16* pwv2   = takeB((size_t)64 * 64);
  __bf16* pwz    = takeB((size_t)128 * 128);
  __bf16* pwm    = takeB((size_t)192 * 192);
  __bf16* hmid_t = takeB((size_t)4 * HWSZ * 64);
  __bf16* m_t    = takeB((size_t)4 * HWSZ * 64);
  __bf16* Qt     = takeB((size_t)4 * HWSZ * 16);
  __bf16* Kht    = takeB((size_t)4 * HWSZ * 16);
  __bf16* Kmt    = takeB((size_t)4 * HWSZ * 16);
  __bf16* Vhc    = takeB((size_t)4 * 64 * HWSZ);
  __bf16* Vmc    = takeB((size_t)4 * 64 * HWSZ);
  __bf16* Zcat   = takeB((size_t)4 * HWSZ * 128);
  __bf16* cat2   = takeB((size_t)4 * HWSZ * 192);
  __bf16* comb   = takeB((size_t)4 * HWSZ * 192);

  // --- packing / layout passes -------------------------------------------
  k_pack_conv<<<dim3(3200), 256, 0, stream>>>(cw, wpc);
  k_pack_w<<<dim3(4),   256, 0, stream>>>(wq,  pwq,  64);
  k_pack_w<<<dim3(4),   256, 0, stream>>>(wk,  pwk,  64);
  k_pack_w<<<dim3(16),  256, 0, stream>>>(wv,  pwv,  64);
  k_pack_w<<<dim3(4),   256, 0, stream>>>(wk2, pwk2, 64);
  k_pack_w<<<dim3(16),  256, 0, stream>>>(wv2, pwv2, 64);
  k_pack_w<<<dim3(64),  256, 0, stream>>>(wz,  pwz,  128);
  k_pack_w<<<dim3(144), 256, 0, stream>>>(wmw, pwm,  192);
  k_xh_pad<<<dim3(9248), 256, 0, stream>>>(x, h, xhp);
  k_t64<<<dim3(4096), 256, 0, stream>>>(m, m_t);

  // --- main pipeline ------------------------------------------------------
  k_conv_gates<<<dim3(32, 16, 4), 32, 0, stream>>>(xhp, wpc, cb, cc);
  k_groupnorm <<<dim3(1024), 256, 0, stream>>>(cc, gng, gnb);
  k_gates     <<<dim3(NTOT / 256), 256, 0, stream>>>(cc, c, out + (size_t)NTOT,
                                                     hmid_t, cat2);

  const long sx64 = 4096l * 64;
  k_gemm_nm<<<dim3(32, 1, 4), 32, 0, stream>>>(pwq,  bq,  hmid_t, Qt,  64, 16,
                                               sx64, 4096l * 16);
  k_gemm_nm<<<dim3(32, 1, 4), 32, 0, stream>>>(pwk,  bk,  hmid_t, Kht, 64, 16,
                                               sx64, 4096l * 16);
  k_gemm_nm<<<dim3(32, 1, 4), 32, 0, stream>>>(pwk2, bk2, m_t,    Kmt, 64, 16,
                                               sx64, 4096l * 16);
  k_gemm_cm<<<dim3(32, 4, 4), 32, 0, stream>>>(pwv,  bv,  hmid_t, Vhc, 64,
                                               sx64, 64l * 4096);
  k_gemm_cm<<<dim3(32, 4, 4), 32, 0, stream>>>(pwv2, bv2, m_t,    Vmc, 64,
                                               sx64, 64l * 4096);

  k_flash<<<dim3(32, 2, 4), 256, 0, stream>>>(Qt, Kht, Vhc, Kmt, Vmc, Zcat);

  k_gemm_nm<<<dim3(32, 8, 4), 32, 0, stream>>>(pwz, bz, Zcat, cat2, 128, 192,
                                               4096l * 128, 4096l * 192);
  k_gemm_nm<<<dim3(32, 12, 4), 32, 0, stream>>>(pwm, bm, cat2, comb, 192, 192,
                                                4096l * 192, 4096l * 192);
  k_mem_out<<<dim3(NTOT / 256), 256, 0, stream>>>(comb, m, out);
}
